// TransitionDown_29472065585604
// MI455X (gfx1250) — matrix operand order repaired and checked by reading, not compiled, not run on hardware
//
#include <hip/hip_runtime.h>
#include <hip/hip_bf16.h>

// Problem constants (from reference)
#define L_PTS   8192
#define N_B     8
#define C_DIM   3
#define DIM_IN  64
#define DIM_OUT 128
#define K_NN    16
#define L2_PTS  2048
#define LN_EPS  1e-5f

// Padded LDS row strides (bf16 elements); multiples of 8 elems (16B) so b128
// loads stay aligned and 16 consecutive rows hit distinct 16B slots mod 256B.
#define LDW1 (DIM_IN + 8)    // 72 elems = 144 B
#define LDW2 (DIM_OUT + 8)   // 136 elems = 272 B
#define LDH  (DIM_OUT + 8)   // 136 elems = 272 B

typedef __attribute__((ext_vector_type(16))) __bf16 v16bf;
typedef __attribute__((ext_vector_type(8)))  __bf16 v8bf;
typedef __attribute__((ext_vector_type(8)))  float  v8f;
typedef unsigned int v4u __attribute__((ext_vector_type(4)));
typedef int          v4i __attribute__((ext_vector_type(4)));
typedef int          v8i __attribute__((ext_vector_type(8)));

union BFrag { v16bf v; v8bf h[2]; };

#if __has_builtin(__builtin_amdgcn_tensor_load_to_lds)
#define TD_HAVE_TDM 1
#else
#define TD_HAVE_TDM 0
#endif

// ds_swizzle_b32 XOR butterfly within 16 lanes: single DS op, immediate pattern
template <int MASK>
__device__ __forceinline__ float xor_add16(float v) {
    int t = __builtin_amdgcn_ds_swizzle(__float_as_int(v), (MASK << 10) | 0x1F);
    return v + __int_as_float(t);
}
__device__ __forceinline__ float red16(float v) {
    return xor_add16<8>(xor_add16<4>(xor_add16<2>(xor_add16<1>(v))));
}

// ---------------------------------------------------------------------------
// Fused MLP: h = LN(x @ w1^T) ; y = relu(LN(h @ w2^T))
// ---------------------------------------------------------------------------
__global__ __launch_bounds__(256)
void TD_mlp_wmma_kernel(const float* __restrict__ x,
                        const float* __restrict__ w1,
                        const float* __restrict__ g1, const float* __restrict__ b1,
                        const float* __restrict__ w2,
                        const float* __restrict__ g2, const float* __restrict__ b2,
                        float* __restrict__ feats)
{
    __shared__ alignas(16) __bf16 Bs1[DIM_OUT][LDW1];  // w1 row-major, bf16
    __shared__ alignas(16) __bf16 Bs2[DIM_OUT][LDW2];  // w2 row-major, bf16
    __shared__ alignas(16) __bf16 Hn[8][16][LDH];      // per-wave normalized h

    const int tid  = threadIdx.x;
    const int wave = tid >> 5;
    const int lane = tid & 31;

    // ---- stage weights (straight copy, f32 -> bf16) ----
    for (int i = tid; i < DIM_OUT * DIM_IN; i += 256) {
        Bs1[i / DIM_IN][i % DIM_IN] = (__bf16)w1[i];
    }
    for (int i = tid; i < DIM_OUT * DIM_OUT; i += 256) {
        Bs2[i / DIM_OUT][i % DIM_OUT] = (__bf16)w2[i];
    }
    __syncthreads();

    const int rowBase = blockIdx.x * 128 + wave * 16;
    const int mrow  = lane & 15;
    const int koff  = (lane >> 4) * 8;
    const int ncol  = lane & 15;
    const int kg    = (lane >> 4) * 8;
    const int rhalf = (lane >> 4) * 8;

    // ---- A fragments for GEMM1 (K=64 -> 2 frags), b128 loads + pk-cvt ----
    const float4* xr4 = (const float4*)(x + (size_t)(rowBase + mrow) * DIM_IN);
    const int q0 = koff >> 2;
    v16bf a0, a1;
    {
        float4 r0 = xr4[q0],      r1 = xr4[q0 + 1];
        float4 r2 = xr4[q0 + 4],  r3 = xr4[q0 + 5];
        float4 r4 = xr4[q0 + 8],  r5 = xr4[q0 + 9];
        float4 r6 = xr4[q0 + 12], r7 = xr4[q0 + 13];
        a0[0]=(__bf16)r0.x; a0[1]=(__bf16)r0.y; a0[2]=(__bf16)r0.z; a0[3]=(__bf16)r0.w;
        a0[4]=(__bf16)r1.x; a0[5]=(__bf16)r1.y; a0[6]=(__bf16)r1.z; a0[7]=(__bf16)r1.w;
        a0[8]=(__bf16)r2.x; a0[9]=(__bf16)r2.y; a0[10]=(__bf16)r2.z; a0[11]=(__bf16)r2.w;
        a0[12]=(__bf16)r3.x; a0[13]=(__bf16)r3.y; a0[14]=(__bf16)r3.z; a0[15]=(__bf16)r3.w;
        a1[0]=(__bf16)r4.x; a1[1]=(__bf16)r4.y; a1[2]=(__bf16)r4.z; a1[3]=(__bf16)r4.w;
        a1[4]=(__bf16)r5.x; a1[5]=(__bf16)r5.y; a1[6]=(__bf16)r5.z; a1[7]=(__bf16)r5.w;
        a1[8]=(__bf16)r6.x; a1[9]=(__bf16)r6.y; a1[10]=(__bf16)r6.z; a1[11]=(__bf16)r6.w;
        a1[12]=(__bf16)r7.x; a1[13]=(__bf16)r7.y; a1[14]=(__bf16)r7.z; a1[15]=(__bf16)r7.w;
    }

    // ---- GEMM1: 8 N-tiles x 2 K-steps; B frags = contiguous b128 pairs ----
    v8f acc[8];
#pragma unroll
    for (int nt = 0; nt < 8; ++nt) {
        const __bf16* bp = &Bs1[nt * 16 + ncol][kg];
        BFrag b0, b1;
        b0.h[0] = *(const v8bf*)(bp);
        b0.h[1] = *(const v8bf*)(bp + 16);
        b1.h[0] = *(const v8bf*)(bp + 32);
        b1.h[1] = *(const v8bf*)(bp + 48);
        v8f c = {};
        c = __builtin_amdgcn_wmma_f32_16x16x32_bf16(false, a0, false, b0.v, (short)0, c, false, false);
        c = __builtin_amdgcn_wmma_f32_16x16x32_bf16(false, a1, false, b1.v, (short)0, c, false, false);
        acc[nt] = c;
    }

    // ---- LayerNorm 1: per-row stats via ds_swizzle XOR reduction ----
    float s[8], sq[8];
#pragma unroll
    for (int e = 0; e < 8; ++e) { s[e] = 0.f; sq[e] = 0.f; }
#pragma unroll
    for (int nt = 0; nt < 8; ++nt)
#pragma unroll
        for (int e = 0; e < 8; ++e) { float v = acc[nt][e]; s[e] += v; sq[e] += v * v; }
#pragma unroll
    for (int e = 0; e < 8; ++e) { s[e] = red16(s[e]); sq[e] = red16(sq[e]); }

    float gc[8], bc[8];
#pragma unroll
    for (int nt = 0; nt < 8; ++nt) { gc[nt] = g1[nt * 16 + ncol]; bc[nt] = b1[nt * 16 + ncol]; }
#pragma unroll
    for (int e = 0; e < 8; ++e) {
        float mean = s[e] * (1.f / 128.f);
        float var  = sq[e] * (1.f / 128.f) - mean * mean;
        float rs   = rsqrtf(var + LN_EPS);
#pragma unroll
        for (int nt = 0; nt < 8; ++nt) {
            float v = (acc[nt][e] - mean) * rs * gc[nt] + bc[nt];
            Hn[wave][rhalf + e][nt * 16 + ncol] = (__bf16)v;
        }
    }

    // ---- A fragments for GEMM2 (K=128 -> 4 frags): 2x b128 LDS loads each
    BFrag a2[4];
#pragma unroll
    for (int f = 0; f < 4; ++f) {
        const __bf16* hp = &Hn[wave][mrow][f * 32 + koff];
        a2[f].h[0] = *(const v8bf*)(hp);
        a2[f].h[1] = *(const v8bf*)(hp + 16);
    }

    // ---- GEMM2: 8 N-tiles x 4 K-steps ----
    v8f acc2[8];
#pragma unroll
    for (int nt = 0; nt < 8; ++nt) {
        const __bf16* bp = &Bs2[nt * 16 + ncol][kg];
        v8f c = {};
#pragma unroll
        for (int f = 0; f < 4; ++f) {
            BFrag bf;
            bf.h[0] = *(const v8bf*)(bp + f * 32);
            bf.h[1] = *(const v8bf*)(bp + f * 32 + 16);
            c = __builtin_amdgcn_wmma_f32_16x16x32_bf16(false, a2[f].v, false, bf.v, (short)0, c, false, false);
        }
        acc2[nt] = c;
    }

    // ---- LayerNorm 2 + ReLU + store ----
#pragma unroll
    for (int e = 0; e < 8; ++e) { s[e] = 0.f; sq[e] = 0.f; }
#pragma unroll
    for (int nt = 0; nt < 8; ++nt)
#pragma unroll
        for (int e = 0; e < 8; ++e) { float v = acc2[nt][e]; s[e] += v; sq[e] += v * v; }
#pragma unroll
    for (int e = 0; e < 8; ++e) { s[e] = red16(s[e]); sq[e] = red16(sq[e]); }

#pragma unroll
    for (int nt = 0; nt < 8; ++nt) { gc[nt] = g2[nt * 16 + ncol]; bc[nt] = b2[nt * 16 + ncol]; }
#pragma unroll
    for (int e = 0; e < 8; ++e) {
        float mean = s[e] * (1.f / 128.f);
        float var  = sq[e] * (1.f / 128.f) - mean * mean;
        float rs   = rsqrtf(var + LN_EPS);
        float* orow = feats + (size_t)(rowBase + rhalf + e) * DIM_OUT;
#pragma unroll
        for (int nt = 0; nt < 8; ++nt) {
            float v = (acc2[nt][e] - mean) * rs * gc[nt] + bc[nt];
            orow[nt * 16 + ncol] = fmaxf(v, 0.f);
        }
    }
}

#if TD_HAVE_TDM
// Issue a TDM load of a 3x256 f32 tile (row stride N_B*3 elems) into LDS.
// D# built per CDNA5 ISA ch.8: group0 = {count/lds/global/type}, group1 = dims.
// amdgpu-toolchain builtin form: (v4u g0, v8i g1, v4i g2, v4i g3, v8i g4, cpol)
__device__ __forceinline__ void tdm_load_coords(const float* gptr, unsigned lds_off) {
    unsigned long long ga = (unsigned long long)(uintptr_t)gptr;
    v4u g0;
    g0[0] = 1u;                                          // count=1, no gather
    g0[1] = lds_off;                                     // lds_addr (bytes)
    g0[2] = (unsigned)(ga & 0xFFFFFFFFu);                // global_addr lo
    g0[3] = (unsigned)((ga >> 32) & 0x01FFFFFFu) | (2u << 30);  // addr hi | type=2
    v8i g1;
    g1[0] = (int)(2u << 16);                // wg_mask=0, data_size=2 (4B)
    g1[1] = (int)(3u << 16);                // tensor_dim0 = 3 (lo16)
    g1[2] = 0;                              // tensor_dim0 hi, tensor_dim1 lo16 = 0
    g1[3] = (int)(0x10u | (3u << 16));      // tensor_dim1 hi16 (dim1=2^20), tile_dim0=3
    g1[4] = 256;                            // tile_dim1=256, tile_dim2=0
    g1[5] = (int)(N_B * C_DIM);             // tensor_dim0_stride = 24 (lo32)
    g1[6] = 0;                              // stride hi, dim1_stride lo
    g1[7] = 0;
    v4i z4 = {0, 0, 0, 0};
    v8i z8 = {0, 0, 0, 0, 0, 0, 0, 0};
    __builtin_amdgcn_tensor_load_to_lds(g0, g1, z4, z4, z8, 0);
}
#endif

// ---------------------------------------------------------------------------
// KNN: one thread per (query q, batch n); candidate coords double-buffered in
// LDS via the Tensor Data Mover (async, overlapped with top-K work).
// ---------------------------------------------------------------------------
__global__ __launch_bounds__(256)
void TD_knn_kernel(const float* __restrict__ coords,   // [L, N, 3]
                   const int* __restrict__ keep_idx,   // [L2]
                   float* __restrict__ keep_out,       // [L2, N, 3]
                   int* __restrict__ nn_out)           // [N, L2, 16]
{
    __shared__ alignas(16) float cs[2][256 * 3];
    const int tid = threadIdx.x;
    const int q = blockIdx.x * 256 + tid;
    const int n = blockIdx.y;

    const int ki = keep_idx[q];
    const float qx = coords[((size_t)ki * N_B + n) * 3 + 0];
    const float qy = coords[((size_t)ki * N_B + n) * 3 + 1];
    const float qz = coords[((size_t)ki * N_B + n) * 3 + 2];
    keep_out[((size_t)q * N_B + n) * 3 + 0] = qx;
    keep_out[((size_t)q * N_B + n) * 3 + 1] = qy;
    keep_out[((size_t)q * N_B + n) * 3 + 2] = qz;

    float bd[K_NN]; int bi[K_NN];
#pragma unroll
    for (int t = 0; t < K_NN; ++t) { bd[t] = 3.4e38f; bi[t] = 0; }

    int cur = 0;
#if TD_HAVE_TDM
    const unsigned loff0 = (unsigned)(uintptr_t)(__attribute__((address_space(3))) float*)&cs[0][0];
    const unsigned loff1 = (unsigned)(uintptr_t)(__attribute__((address_space(3))) float*)&cs[1][0];
    const float* gbase = coords + (size_t)n * C_DIM;   // p=0 row for batch n
    if (tid < 32) tdm_load_coords(gbase, loff0);       // prologue: tile 0
#endif

    for (int tile = 0; tile < L_PTS; tile += 256) {
#if TD_HAVE_TDM
        if (tid < 32) {
            if (tile + 256 < L_PTS) {
                tdm_load_coords(gbase + (size_t)(tile + 256) * N_B * C_DIM,
                                cur ? loff0 : loff1);
                __builtin_amdgcn_s_wait_tensorcnt(1);  // current tile done
            } else {
                __builtin_amdgcn_s_wait_tensorcnt(0);
            }
        }
        __syncthreads();
#else
        __syncthreads();
        const int p = tile + tid;
        cs[cur][tid * 3 + 0] = coords[((size_t)p * N_B + n) * 3 + 0];
        cs[cur][tid * 3 + 1] = coords[((size_t)p * N_B + n) * 3 + 1];
        cs[cur][tid * 3 + 2] = coords[((size_t)p * N_B + n) * 3 + 2];
        if (tile + 256 < L_PTS)
            __builtin_prefetch(&coords[((size_t)(p + 256) * N_B + n) * 3], 0, 0);
        __syncthreads();
#endif
        const float* cc = cs[cur];
        for (int j = 0; j < 256; ++j) {
            const float dx = qx - cc[j * 3 + 0];
            const float dy = qy - cc[j * 3 + 1];
            const float dz = qz - cc[j * 3 + 2];
            const float d = dx * dx + dy * dy + dz * dz;
            if (d < bd[K_NN - 1]) {
                bd[K_NN - 1] = d; bi[K_NN - 1] = tile + j;
#pragma unroll
                for (int t = K_NN - 1; t > 0; --t) {
                    if (bd[t] < bd[t - 1]) {
                        float td = bd[t]; bd[t] = bd[t - 1]; bd[t - 1] = td;
                        int   ti = bi[t]; bi[t] = bi[t - 1]; bi[t - 1] = ti;
                    }
                }
            }
        }
#if TD_HAVE_TDM
        __syncthreads();   // all waves done with cs[cur] before TDM reuses it
        cur ^= 1;
#endif
    }
    int* o = nn_out + ((size_t)n * L2_PTS + q) * K_NN;
#pragma unroll
    for (int t = 0; t < K_NN; ++t) o[t] = bi[t] & (L2_PTS - 1);
}

// ---------------------------------------------------------------------------
// Gather + max-pool over K neighbors: one block per (q, n), thread = feat dim
// ---------------------------------------------------------------------------
__global__ __launch_bounds__(128)
void TD_pool_kernel(const float* __restrict__ feats,  // [L*N, 128]
                    const int* __restrict__ nn,       // [N, L2, 16]
                    float* __restrict__ out)          // [L2, N, 128]
{
    const int q = blockIdx.x;
    const int n = blockIdx.y;
    const int d = threadIdx.x;
    const int* row = nn + ((size_t)n * L2_PTS + q) * K_NN;

#pragma unroll
    for (int k = 0; k < K_NN; ++k)
        __builtin_prefetch(&feats[((size_t)row[k] * N_B + n) * DIM_OUT + d], 0, 0);

    float m = -3.4e38f;
#pragma unroll
    for (int k = 0; k < K_NN; ++k) {
        const int id = row[k];
        const float v = feats[((size_t)id * N_B + n) * DIM_OUT + d];
        m = fmaxf(m, v);
    }
    out[((size_t)q * N_B + n) * DIM_OUT + d] = m;
}

extern "C" void kernel_launch(void* const* d_in, const int* in_sizes, int n_in,
                              void* d_out, int out_size, void* d_ws, size_t ws_size,
                              hipStream_t stream) {
    (void)in_sizes; (void)n_in; (void)out_size; (void)ws_size;
    const float* coords   = (const float*)d_in[0];
    const float* features = (const float*)d_in[1];
    const float* w1       = (const float*)d_in[2];
    const float* g1       = (const float*)d_in[3];
    const float* b1       = (const float*)d_in[4];
    const float* w2       = (const float*)d_in[5];
    const float* g2       = (const float*)d_in[6];
    const float* b2       = (const float*)d_in[7];
    const int*   keep_idx = (const int*)d_in[8];

    float* out      = (float*)d_out;
    float* keep_out = out;                                // [L2, N, 3]
    float* pool_out = out + (size_t)L2_PTS * N_B * C_DIM; // [L2, N, 128]

    float* feats_ws = (float*)d_ws;                       // [L*N, 128] f32
    int*   nn_ws    = (int*)((char*)d_ws + (size_t)L_PTS * N_B * DIM_OUT * sizeof(float));

    TD_mlp_wmma_kernel<<<512, 256, 0, stream>>>(features, w1, g1, b1, w2, g2, b2, feats_ws);
    TD_knn_kernel<<<dim3(L2_PTS / 256, N_B), 256, 0, stream>>>(coords, keep_idx, keep_out, nn_ws);
    TD_pool_kernel<<<dim3(L2_PTS, N_B), 128, 0, stream>>>(feats_ws, nn_ws, pool_out);
}
